// CustomLSTMModel_11742440587680
// MI455X (gfx1250) — matrix operand order repaired and checked by reading, not compiled, not run on hardware
//
#include <hip/hip_runtime.h>
#include <hip/hip_bf16.h>

// ---------------- problem constants ----------------
#define B_   32
#define T_   2048
#define D_   512
#define H_   512
#define G4H  2048      // 4*H concatenated gate columns (i,f,o,c)
#define NWG_SCAN 32    // persistent workgroups in recurrent scan
#define WP   520       // padded LDS pitch (elements) for Wh slab (kills bank conflicts)
#define HP   520       // padded LDS pitch (elements) for h slab

typedef __bf16 bf16;
typedef __attribute__((ext_vector_type(16))) __bf16 bf16x16;
typedef __attribute__((ext_vector_type(8)))  float  floatx8;

union V16 { bf16x16 v; uint4 q[2]; };

// A-fragment (16x32 bf16): lane L -> row M = L%16;
// elems 0..15 = K = (L/16)*16 .. +15 (contiguous, matches B)
__device__ __forceinline__ bf16x16 load_a_frag(const bf16* Atile, int lda, int lane) {
  const int m  = lane & 15;
  const int kh = lane >> 4;
  const bf16* p = Atile + (size_t)m * lda + kh * 16;
  V16 u;
  u.q[0] = *(const uint4*)(p);
  u.q[1] = *(const uint4*)(p + 8);
  return u.v;
}

// B-fragment (32x16 bf16): lane L -> col N = L%16, 16 contiguous K at (L/16)*16.
// Weights stored TRANSPOSED ([N][K]) so this is 32 contiguous bytes.
__device__ __forceinline__ bf16x16 load_b_frag(const bf16* WTtile, int ldk, int lane) {
  const int n  = lane & 15;
  const int kh = lane >> 4;
  const bf16* p = WTtile + (size_t)n * ldk + kh * 16;
  V16 u;
  u.q[0] = *(const uint4*)(p);
  u.q[1] = *(const uint4*)(p + 8);
  return u.v;
}

__device__ __forceinline__ floatx8 wmma_bf16(bf16x16 a, bf16x16 b, floatx8 c) {
  return __builtin_amdgcn_wmma_f32_16x16x32_bf16(
      /*neg_a=*/false, a, /*neg_b=*/false, b,
      /*c_mod=*/(short)0, c, /*reuse_a=*/false, /*reuse_b=*/false);
}

// ---------------- phase 0a: x fp32 -> bf16 ----------------
__global__ void k_convert_x(const float* __restrict__ x, bf16* __restrict__ xbf, int n4) {
  union U { bf16 h[4]; uint2 u; };
  int i = blockIdx.x * blockDim.x + threadIdx.x;
  const int stride = gridDim.x * blockDim.x;
  for (; i < n4; i += stride) {
    float4 f = ((const float4*)x)[i];
    U o;
    o.h[0] = (bf16)f.x; o.h[1] = (bf16)f.y; o.h[2] = (bf16)f.z; o.h[3] = (bf16)f.w;
    ((uint2*)xbf)[i] = o.u;
  }
}

// ---------------- phase 0b: pack W^T (bf16) + bias concat ----------------
__global__ void k_pack_w(const float* __restrict__ Wxi, const float* __restrict__ Whi, const float* __restrict__ bi,
                         const float* __restrict__ Wxf, const float* __restrict__ Whf, const float* __restrict__ bfv,
                         const float* __restrict__ Wxo, const float* __restrict__ Who, const float* __restrict__ bo,
                         const float* __restrict__ Wxc, const float* __restrict__ Whc, const float* __restrict__ bc,
                         bf16* __restrict__ WxT, bf16* __restrict__ WhT, float* __restrict__ bcat) {
  const int n = blockIdx.x;          // 0..2047 concatenated gate column
  const int g = n >> 9;              // gate: 0=i 1=f 2=o 3=c
  const int j = n & 511;             // column within gate
  const float* Wx = (g == 0) ? Wxi : (g == 1) ? Wxf : (g == 2) ? Wxo : Wxc;
  const float* Wh = (g == 0) ? Whi : (g == 1) ? Whf : (g == 2) ? Who : Whc;
  const float* bb = (g == 0) ? bi  : (g == 1) ? bfv : (g == 2) ? bo  : bc;
  for (int k = threadIdx.x; k < 512; k += blockDim.x) {
    WxT[(size_t)n * 512 + k] = (bf16)Wx[(size_t)k * 512 + j];
    WhT[(size_t)n * 512 + k] = (bf16)Wh[(size_t)k * 512 + j];
  }
  if (threadIdx.x == 0) bcat[n] = bb[j];
}

// ---------------- phase 0c: zero h double-buffer + grid-barrier counter ----------------
__global__ void k_zero(uint4* __restrict__ hbuf, unsigned* __restrict__ ctr) {
  const int i = blockIdx.x * blockDim.x + threadIdx.x;
  if (i < (2 * B_ * H_ * 2 / 16)) hbuf[i] = make_uint4(0u, 0u, 0u, 0u);
  if (i == 0) *ctr = 0u;
}

// ---------------- phase 1: gates = x @ WxCat + b  (M=65536, K=512, N=2048) --------
// 512 blocks x 8 waves; each wave owns one 16-row M tile, sweeps N in 64-wide chunks.
// Fully-unrolled SSA K-loop (no back-edge register copies) + sched_group_barrier
// interleave hints. The A offset is laundered through a VGPR asm each nc iteration
// (it is divergent — derived from threadIdx — so an SGPR constraint is illegal):
// base-pointer provenance is preserved (global_load, not flat_load) while LLVM
// cannot hoist all 16 A fragments out of the loop (previous spill cause).
__global__ void k_gates_gemm(const bf16* __restrict__ xbf, const bf16* __restrict__ WxT,
                             const float* __restrict__ bcat, bf16* __restrict__ gates) {
  const int lane  = threadIdx.x & 31;
  const int wave  = threadIdx.x >> 5;
  const int mtile = blockIdx.x * 8 + wave;
  const size_t row0 = (size_t)mtile * 16;
  const floatx8 zero = {0.f, 0.f, 0.f, 0.f, 0.f, 0.f, 0.f, 0.f};

#pragma unroll 1
  for (int nc = 0; nc < 32; ++nc) {
    size_t aoff = row0 * D_;
    asm("" : "+v"(aoff));                      // opaque divergent offset: no cross-nc hoist
    const bf16* Arow  = xbf + aoff;
    const bf16* Bbase = WxT + (size_t)nc * 64 * D_;

    floatx8 acc[4];
#pragma unroll
    for (int i = 0; i < 4; ++i) acc[i] = zero;

#pragma unroll
    for (int kk = 0; kk < 16; ++kk) {
      bf16x16 a  = load_a_frag(Arow + kk * 32, D_, lane);
      bf16x16 q0 = load_b_frag(Bbase + (size_t)0 * 16 * D_ + kk * 32, D_, lane);
      bf16x16 q1 = load_b_frag(Bbase + (size_t)1 * 16 * D_ + kk * 32, D_, lane);
      bf16x16 q2 = load_b_frag(Bbase + (size_t)2 * 16 * D_ + kk * 32, D_, lane);
      bf16x16 q3 = load_b_frag(Bbase + (size_t)3 * 16 * D_ + kk * 32, D_, lane);
      acc[0] = wmma_bf16(a, q0, acc[0]);
      acc[1] = wmma_bf16(a, q1, acc[1]);
      acc[2] = wmma_bf16(a, q2, acc[2]);
      acc[3] = wmma_bf16(a, q3, acc[3]);
#if __has_builtin(__builtin_amdgcn_sched_group_barrier)
      // per K-step: 10 VMEM reads + 4 WMMAs, interleaved in two half-groups
      __builtin_amdgcn_sched_group_barrier(0x020, 5, 0);  // 5 VMEM reads
      __builtin_amdgcn_sched_group_barrier(0x008, 2, 0);  // 2 WMMA
      __builtin_amdgcn_sched_group_barrier(0x020, 5, 0);
      __builtin_amdgcn_sched_group_barrier(0x008, 2, 0);
#endif
    }

    // epilogue: + bias, fp32 -> bf16, store (C/D layout: lane -> N=L%16, M=r+(L/16)*8)
    const int nl = lane & 15;
    const int mb = (lane >> 4) * 8;
#pragma unroll
    for (int nt = 0; nt < 4; ++nt) {
      const int n = nc * 64 + nt * 16 + nl;
      const float bias = bcat[n];
#pragma unroll
      for (int r = 0; r < 8; ++r) {
        const size_t m = row0 + mb + r;
        gates[m * G4H + n] = (bf16)(acc[nt][r] + bias);
      }
    }
  }
}

// ---------------- phase 2: persistent recurrent scan -------------------------------
// 32 WGs x 256 threads. WG wg owns H-columns [wg*16, wg*16+16) for ALL 4 gates.
// Wh slice (4*16 cols x 512 K bf16) lives in LDS (padded pitch) for the whole scan.
// Each step, h (32x512 bf16) is pulled into LDS with global_load_async_to_lds_b128
// (ASYNCcnt) so WMMA fragments come from conflict-free ds_loads. Cell state c lives
// in LDS. Cross-WG sync = monotonically increasing atomic grid barrier per step.
__global__ void k_lstm_scan(const bf16* __restrict__ gates, const bf16* __restrict__ WhT,
                            bf16* __restrict__ hbuf, bf16* __restrict__ hs,
                            unsigned* __restrict__ ctr) {
  extern __shared__ char smem[];
  bf16*  wh    = (bf16*)smem;                                   // [64][WP] bf16
  bf16*  hsl   = (bf16*)(smem + 64 * WP * 2);                   // [32][HP] bf16
  float* gbuf  = (float*)(smem + 64 * WP * 2 + 32 * HP * 2);    // [4][32][16] f32
  float* cslab = gbuf + 4 * 32 * 16;                            // [32][16]    f32

  const int wg   = blockIdx.x;
  const int c0   = wg * 16;
  const int lane = threadIdx.x & 31;
  const int wave = threadIdx.x >> 5;
  const int g    = wave >> 1;      // gate 0..3
  const int mt   = wave & 1;       // M tile (batch rows 0-15 / 16-31)

  // preload Wh^T slice into padded LDS slab: row (g*16+j) <- WhT row (g*512+c0+j)
  for (int i = threadIdx.x; i < 64 * 512 / 8; i += blockDim.x) {
    const int e = i * 8;
    const int rrow = e >> 9;
    const int k = e & 511;
    const int gg = rrow >> 4, jj = rrow & 15;
    *(uint4*)(wh + (size_t)rrow * WP + k) =
        *(const uint4*)(WhT + ((size_t)(gg * 512 + c0 + jj)) * 512 + k);
  }
  for (int i = threadIdx.x; i < 512; i += blockDim.x) cslab[i] = 0.f;
  __syncthreads();

  const int nl  = lane & 15;
  const int mb  = mt * 16 + (lane >> 4) * 8;
  const int col = g * 512 + c0 + nl;
  const unsigned hsl_base = (unsigned)(uintptr_t)(void*)hsl;
  const bf16* Asl = hsl + mt * 16 * HP;          // this wave's A rows in LDS
  const bf16* Bsl = wh + (size_t)g * 16 * WP;    // this wave's Wh cols in LDS

  // prefetch t=0 gate pre-activations (independent of h)
  float gv[8];
#pragma unroll
  for (int r = 0; r < 8; ++r)
    gv[r] = (float)gates[((size_t)(mb + r) * T_) * G4H + col];

  for (int t = 0; t < T_; ++t) {
    // ---- stage h[t] into LDS via CDNA5 async copy (32 rows x 512 bf16) ----
    {
      const unsigned long long gbase =
          (unsigned long long)(uintptr_t)(hbuf + (size_t)(t & 1) * (B_ * H_));
#pragma unroll
      for (int i = 0; i < 8; ++i) {
        const int c = threadIdx.x + i * 256;        // 2048 chunks of 16B
        const int row = c >> 6, col8 = c & 63;
        const unsigned ldsoff = hsl_base + (unsigned)(row * HP + col8 * 8) * 2u;
        const unsigned goff   = (unsigned)(row * 512 + col8 * 8) * 2u;
        asm volatile("global_load_async_to_lds_b128 %0, %1, %2"
                     :: "v"(ldsoff), "v"(goff), "s"(gbase) : "memory");
      }
      asm volatile("s_wait_asynccnt 0x0" ::: "memory");
    }
    __syncthreads();

    // ---- C init = precomputed x-projection (already biased) ----
    floatx8 acc;
#pragma unroll
    for (int r = 0; r < 8; ++r) acc[r] = gv[r];

    // ---- h @ Wh slice: 16 WMMAs over K=512, SSA fragments from LDS ----
#pragma unroll
    for (int kk = 0; kk < 16; ++kk) {
      bf16x16 a = load_a_frag(Asl + kk * 32, HP, lane);
      bf16x16 b = load_b_frag(Bsl + kk * 32, WP, lane);
      acc = wmma_bf16(a, b, acc);
#if __has_builtin(__builtin_amdgcn_sched_group_barrier)
      __builtin_amdgcn_sched_group_barrier(0x100, 4, 0);  // 4 DS reads
      __builtin_amdgcn_sched_group_barrier(0x008, 1, 0);  // 1 WMMA
#endif
    }

    // prefetch next timestep's gate pre-activations (latency hidden by barrier)
    if (t + 1 < T_) {
#pragma unroll
      for (int r = 0; r < 8; ++r)
        gv[r] = (float)gates[((size_t)(mb + r) * T_ + (t + 1)) * G4H + col];
    }

    // ---- per-gate activation, stage fp32 tile in LDS ----
#pragma unroll
    for (int r = 0; r < 8; ++r) {
      float v = acc[r];
      v = (g == 3) ? tanhf(v) : (1.0f / (1.0f + __expf(-v)));
      gbuf[(g * 32 + mb + r) * 16 + nl] = v;
    }
    __syncthreads();

    // ---- elementwise cell update for the owned 32x16 slice ----
    for (int e = threadIdx.x; e < 512; e += blockDim.x) {
      const int b = e >> 4, j = e & 15;
      const float I  = gbuf[(0 * 32 + b) * 16 + j];
      const float F  = gbuf[(1 * 32 + b) * 16 + j];
      const float O  = gbuf[(2 * 32 + b) * 16 + j];
      const float Cg = gbuf[(3 * 32 + b) * 16 + j];
      const float cn = F * cslab[e] + I * Cg;
      cslab[e] = cn;
      const bf16 hn = (bf16)(O * tanhf(cn));
      hbuf[(size_t)((t + 1) & 1) * (B_ * H_) + (size_t)b * H_ + c0 + j] = hn;
      hs[((size_t)b * T_ + t) * H_ + c0 + j] = hn;
    }
    __threadfence();
    __syncthreads();

    // ---- grid barrier: monotonically increasing counter ----
    if (threadIdx.x == 0) {
      __hip_atomic_fetch_add(ctr, 1u, __ATOMIC_ACQ_REL, __HIP_MEMORY_SCOPE_AGENT);
      const unsigned target = (unsigned)NWG_SCAN * (unsigned)(t + 1);
      while (__hip_atomic_load(ctr, __ATOMIC_ACQUIRE, __HIP_MEMORY_SCOPE_AGENT) < target) {
        __builtin_amdgcn_s_sleep(8);
      }
    }
    __syncthreads();
    __threadfence();
  }
}

// ---------------- phase 3: out = hs @ Wd + bd (one wave per row) --------------------
__global__ void k_proj(const bf16* __restrict__ hs, const float* __restrict__ Wd,
                       const float* __restrict__ bd, float* __restrict__ out) {
  const int lane = threadIdx.x & 31;
  const int wave = threadIdx.x >> 5;
  const size_t row = (size_t)blockIdx.x * 8 + wave;
  const bf16* p = hs + row * H_ + lane * 16;
  const float* w = Wd + lane * 16;
  float s = 0.f;
#pragma unroll
  for (int i = 0; i < 16; ++i) s += (float)p[i] * w[i];
#pragma unroll
  for (int off = 16; off >= 1; off >>= 1) s += __shfl_xor(s, off, 32);
  if (lane == 0) out[row] = s + bd[0];
}

// ---------------- host side ----------------
extern "C" void kernel_launch(void* const* d_in, const int* in_sizes, int n_in,
                              void* d_out, int out_size, void* d_ws, size_t ws_size,
                              hipStream_t stream) {
  (void)in_sizes; (void)n_in; (void)out_size; (void)ws_size;
  const float* x   = (const float*)d_in[0];
  const float* Wxi = (const float*)d_in[1];
  const float* Whi = (const float*)d_in[2];
  const float* bi  = (const float*)d_in[3];
  const float* Wxf = (const float*)d_in[4];
  const float* Whf = (const float*)d_in[5];
  const float* bfv = (const float*)d_in[6];
  const float* Wxo = (const float*)d_in[7];
  const float* Who = (const float*)d_in[8];
  const float* bo  = (const float*)d_in[9];
  const float* Wxc = (const float*)d_in[10];
  const float* Whc = (const float*)d_in[11];
  const float* bc  = (const float*)d_in[12];
  const float* Wd  = (const float*)d_in[13];
  const float* bd  = (const float*)d_in[14];
  float* out = (float*)d_out;

  char* ws = (char*)d_ws;
  const size_t OFF_XBF   = 0;                                  // B*T*D bf16   = 64MB
  const size_t OFF_WXT   = OFF_XBF   + (size_t)B_*T_*D_*2;     // 2048x512 bf16 = 2MB
  const size_t OFF_WHT   = OFF_WXT   + (size_t)G4H*D_*2;       // 2048x512 bf16 = 2MB
  const size_t OFF_BC    = OFF_WHT   + (size_t)G4H*H_*2;       // 2048 f32
  const size_t OFF_GATES = OFF_BC    + (size_t)G4H*4;          // 65536x2048 bf16 = 256MB
  const size_t OFF_HBUF  = OFF_GATES + (size_t)B_*T_*G4H*2;    // 2x32x512 bf16
  const size_t OFF_HS    = OFF_HBUF  + (size_t)2*B_*H_*2;      // 65536x512 bf16 = 64MB
  const size_t OFF_CTR   = OFF_HS    + (size_t)B_*T_*H_*2;     // 256B

  bf16*     xbf   = (bf16*)(ws + OFF_XBF);
  bf16*     WxT   = (bf16*)(ws + OFF_WXT);
  bf16*     WhT   = (bf16*)(ws + OFF_WHT);
  float*    bcat  = (float*)(ws + OFF_BC);
  bf16*     gates = (bf16*)(ws + OFF_GATES);
  bf16*     hbuf  = (bf16*)(ws + OFF_HBUF);
  bf16*     hs    = (bf16*)(ws + OFF_HS);
  unsigned* ctr   = (unsigned*)(ws + OFF_CTR);

  // phase 0
  k_convert_x<<<2048, 256, 0, stream>>>(x, xbf, (int)((size_t)B_*T_*D_/4));
  k_pack_w<<<G4H, 256, 0, stream>>>(Wxi, Whi, bi, Wxf, Whf, bfv, Wxo, Who, bo,
                                    Wxc, Whc, bc, WxT, WhT, bcat);
  k_zero<<<16, 256, 0, stream>>>((uint4*)hbuf, ctr);
  // phase 1: big pipelined WMMA GEMM for all 4 input projections
  k_gates_gemm<<<512, 256, 0, stream>>>(xbf, WxT, bcat, gates);
  // phase 2: persistent WMMA recurrent scan (LDS-resident Wh, async h staging)
  const size_t smem = (size_t)64 * WP * 2 + (size_t)32 * HP * 2
                    + 4 * 32 * 16 * 4 + 512 * 4;   // 110080 B
  k_lstm_scan<<<NWG_SCAN, 256, smem, stream>>>(gates, WhT, hbuf, hs, ctr);
  // phase 3: output projection
  k_proj<<<(B_ * T_) / 8, 256, 0, stream>>>(hs, Wd, bd, out);
}
